// MultiHeadTemporalAttention_60799557042416
// MI455X (gfx1250) — matrix operand order repaired and checked by reading, not compiled, not run on hardware
//
#include <hip/hip_runtime.h>

typedef _Float16 h16;
typedef __attribute__((ext_vector_type(16))) _Float16 v16h;
typedef __attribute__((ext_vector_type(8)))  _Float16 v8h;
typedef __attribute__((ext_vector_type(8)))  float    v8f;

#define D_MODEL   1024
#define NUM_HEADS 16
#define D_K       64
#define BATCH     16
#define SEQ       512
#define MROWS     (BATCH * SEQ)          // 8192

#define SC_PITCH  516                    // fp32 score row pitch (bank-conflict-free halves)
#define AT_PITCH  520                    // f16 attn row pitch (16B-aligned rows, conflict-free b128 reads)

// ---------------------------------------------------------------------------
// fp32 -> f16 conversion (grid-stride)
// ---------------------------------------------------------------------------
__global__ void cvt_f32_to_f16(const float* __restrict__ x, h16* __restrict__ y, int n) {
    for (int i = blockIdx.x * blockDim.x + threadIdx.x; i < n; i += gridDim.x * blockDim.x)
        y[i] = (h16)x[i];
}

// ---------------------------------------------------------------------------
// Y = X @ W^T + bias.  X:[Md,Kd] f16 row-major, W:[Nd,Kd] f16 row-major.
// Both operands contract along the contiguous dim -> aligned b128 fragment
// loads.  Block: 8 waves (256 thr) as 2(M) x 4(N) -> 64 x 256 tile.
// Wave: 32(M) x 64(N) = 8 accumulators; per K-step 8 WMMA vs 12 b128 loads
// (1.5 loads/WMMA) with 4-way A reuse and 2-way B reuse.
// Writes f16 (Yh) or fp32 (Yf) selected by non-null pointer.
// ---------------------------------------------------------------------------
__global__ __launch_bounds__(256) void gemm_xwt(
    const h16* __restrict__ X, const h16* __restrict__ W, const float* __restrict__ bias,
    h16* __restrict__ Yh, float* __restrict__ Yf, int Md, int Nd, int Kd)
{
    const int lane = threadIdx.x & 31;
    const int wave = threadIdx.x >> 5;
    const int half = lane >> 4;
    const int l16  = lane & 15;
    const int row0 = blockIdx.x * 64  + (wave & 1) * 32;   // wave: 32 rows
    const int col0 = blockIdx.y * 256 + (wave >> 1) * 64;  // wave: 64 cols

    v8f c[2][4];
#pragma unroll
    for (int i = 0; i < 2; ++i)
#pragma unroll
        for (int j = 0; j < 4; ++j)
            c[i][j] = (v8f){0.f,0.f,0.f,0.f,0.f,0.f,0.f,0.f};

    const h16* Arow0 = X + (size_t)(row0 + l16) * Kd;
    const h16* Arow1 = X + (size_t)(row0 + 16 + l16) * Kd;
    for (int k0 = 0; k0 < Kd; k0 += 32) {
        // A fragments 16x32: lane l16=row; K = 16*half + e
        v16h a[2];
        {
            v8h a00 = *(const v8h*)(Arow0 + k0 + 16 * half);
            v8h a01 = *(const v8h*)(Arow0 + k0 + 16 * half + 8);
            v8h a10 = *(const v8h*)(Arow1 + k0 + 16 * half);
            v8h a11 = *(const v8h*)(Arow1 + k0 + 16 * half + 8);
#pragma unroll
            for (int e = 0; e < 8; ++e) {
                a[0][e] = a00[e]; a[0][e + 8] = a01[e];
                a[1][e] = a10[e]; a[1][e + 8] = a11[e];
            }
        }
#pragma unroll
        for (int j = 0; j < 4; ++j) {
            // B fragment 32x16: lane l16=col n; K = 16*half + e  (contiguous 32B)
            const h16* Brow = W + (size_t)(col0 + j * 16 + l16) * Kd + k0 + 16 * half;
            v16h b = *(const v16h*)Brow;
            c[0][j] = __builtin_amdgcn_wmma_f32_16x16x32_f16(false, a[0], false, b,
                                                             (short)0, c[0][j], false, false);
            c[1][j] = __builtin_amdgcn_wmma_f32_16x16x32_f16(false, a[1], false, b,
                                                             (short)0, c[1][j], false, false);
        }
    }
#pragma unroll
    for (int j = 0; j < 4; ++j) {
        const int n = col0 + j * 16 + l16;
        const float bv = bias[n];
#pragma unroll
        for (int i = 0; i < 2; ++i) {
#pragma unroll
            for (int r = 0; r < 8; ++r) {       // C/D: lane l16=n, m = r + 8*half
                const int m = row0 + i * 16 + 8 * half + r;
                const float v = c[i][j][r] + bv;
                if (Yf) Yf[(size_t)m * Nd + n] = v;
                else    Yh[(size_t)m * Nd + n] = (h16)v;
            }
        }
    }
}

// ---------------------------------------------------------------------------
// Vt[b][h][d][t] = V[b*S+t][h*64+d]   (f16 transpose; V tile lives in L2)
// ---------------------------------------------------------------------------
__global__ void transpose_v(const h16* __restrict__ V, h16* __restrict__ Vt) {
    const size_t N = (size_t)BATCH * NUM_HEADS * D_K * SEQ;
    size_t i = (size_t)blockIdx.x * blockDim.x + threadIdx.x;
    if (i >= N) return;
    const int t = i & (SEQ - 1);
    size_t r = i >> 9;
    const int d = r & (D_K - 1); r >>= 6;
    const int h = r & (NUM_HEADS - 1);
    const int b = (int)(r >> 4);
    Vt[i] = V[((size_t)(b * SEQ + t)) * D_MODEL + h * D_K + d];
}

// ---------------------------------------------------------------------------
// Fused attention: one block per (b, h, 16-row tile).
//  scores(WMMA, K=64) -> +bias, causal mask -> softmax in LDS ->
//  attn f16 to global (for head-mean) -> ctx = attn @ V (WMMA, K=512 split)
// ---------------------------------------------------------------------------
__global__ __launch_bounds__(256) void attn_block(
    const h16* __restrict__ Q, const h16* __restrict__ Km, const h16* __restrict__ Vt,
    const float* __restrict__ rel_bias, h16* __restrict__ attn_out, h16* __restrict__ ctx_out)
{
    __shared__ float sc[16 * SC_PITCH];
    __shared__ __align__(16) h16 ah[16 * AT_PITCH];
    __shared__ float red[4 * 16 * 16];

    const int lane = threadIdx.x & 31;
    const int wave = threadIdx.x >> 5;
    const int half = lane >> 4;
    const int l16  = lane & 15;

    const int mt = blockIdx.x & 31;              // 32 row-tiles of 16
    const int h  = (blockIdx.x >> 5) & (NUM_HEADS - 1);
    const int b  = blockIdx.x >> 9;
    const int s0 = mt * 16;

    // ---- scores: each wave covers t in [wave*64, wave*64+64) -------------
    const h16* Arow = Q + (size_t)(b * SEQ + s0 + l16) * D_MODEL + h * D_K;
    v16h afr[2];
#pragma unroll
    for (int kk = 0; kk < 2; ++kk) {
        v8h a0 = *(const v8h*)(Arow + kk * 32 + 16 * half);
        v8h a1 = *(const v8h*)(Arow + kk * 32 + 16 * half + 8);
#pragma unroll
        for (int e = 0; e < 8; ++e) { afr[kk][e] = a0[e]; afr[kk][e + 8] = a1[e]; }
    }
#pragma unroll
    for (int j = 0; j < 4; ++j) {
        const int t0 = (wave * 4 + j) * 16;
        v8f c = (v8f){0.f,0.f,0.f,0.f,0.f,0.f,0.f,0.f};
#pragma unroll
        for (int kk = 0; kk < 2; ++kk) {
            const h16* Brow = Km + (size_t)(b * SEQ + t0 + l16) * D_MODEL
                                 + h * D_K + kk * 32 + 16 * half;
            v16h bf = *(const v16h*)Brow;
            c = __builtin_amdgcn_wmma_f32_16x16x32_f16(false, afr[kk], false, bf,
                                                       (short)0, c, false, false);
        }
#pragma unroll
        for (int r = 0; r < 8; ++r) {
            const int m = 8 * half + r;
            const int s_idx = s0 + m;
            const int t = t0 + l16;
            float v = c[r] * 0.125f + rel_bias[((size_t)h * SEQ + s_idx) * SEQ + t];
            if (t > s_idx) v = -__builtin_inff();
            sc[m * SC_PITCH + t] = v;
        }
    }
    __syncthreads();

    // ---- softmax: wave handles rows 2*wave, 2*wave+1 ----------------------
    for (int rr = 0; rr < 2; ++rr) {
        const int row = wave * 2 + rr;
        const int s_idx = s0 + row;
        float mx = -__builtin_inff();
        for (int i = lane; i < SEQ; i += 32) mx = fmaxf(mx, sc[row * SC_PITCH + i]);
#pragma unroll
        for (int o = 16; o > 0; o >>= 1) mx = fmaxf(mx, __shfl_xor(mx, o, 32));
        float sum = 0.f;
        for (int i = lane; i < SEQ; i += 32) {
            float e = __expf(sc[row * SC_PITCH + i] - mx);
            sc[row * SC_PITCH + i] = e;
            sum += e;
        }
#pragma unroll
        for (int o = 16; o > 0; o >>= 1) sum += __shfl_xor(sum, o, 32);
        const float inv = 1.f / sum;
        h16* gout = attn_out + ((size_t)(b * NUM_HEADS + h) * SEQ + s_idx) * SEQ;
        for (int i = lane; i < SEQ; i += 32) {
            h16 p = (h16)(sc[row * SC_PITCH + i] * inv);
            ah[row * AT_PITCH + i] = p;
            gout[i] = p;
        }
    }
    __syncthreads();

    // ---- ctx = attn @ V : waves split {4 d-tiles} x {2 K halves} ----------
    const int wn = wave & 3;
    const int kg = wave >> 2;
    v8f c = (v8f){0.f,0.f,0.f,0.f,0.f,0.f,0.f,0.f};
    for (int ks = 0; ks < 8; ++ks) {
        const int k0 = kg * 256 + ks * 32;
        const h16* arow = ah + l16 * AT_PITCH + k0;       // A from LDS (16B aligned)
        v8h a0 = *(const v8h*)(arow + 16 * half);
        v8h a1 = *(const v8h*)(arow + 16 * half + 8);
        v16h a;
#pragma unroll
        for (int e = 0; e < 8; ++e) { a[e] = a0[e]; a[e + 8] = a1[e]; }
        const h16* Brow = Vt + ((size_t)(b * NUM_HEADS + h) * D_K + wn * 16 + l16) * SEQ
                             + k0 + 16 * half;            // contiguous t, 32B aligned
        v16h bf = *(const v16h*)Brow;
        c = __builtin_amdgcn_wmma_f32_16x16x32_f16(false, a, false, bf,
                                                   (short)0, c, false, false);
    }
    if (kg == 1) {
#pragma unroll
        for (int r = 0; r < 8; ++r)
            red[(wn * 16 + 8 * half + r) * 16 + l16] = c[r];
    }
    __syncthreads();
    if (kg == 0) {
#pragma unroll
        for (int r = 0; r < 8; ++r) {
            const int m = 8 * half + r;
            const float v = c[r] + red[(wn * 16 + m) * 16 + l16];
            ctx_out[(size_t)(b * SEQ + s0 + m) * D_MODEL + h * D_K + wn * 16 + l16] = (h16)v;
        }
    }
}

// ---------------------------------------------------------------------------
// attn_mean[b,s,t] = (1/H) * sum_h attn[b,h,s,t]
// ---------------------------------------------------------------------------
__global__ void attn_mean(const h16* __restrict__ attn, float* __restrict__ out) {
    const size_t N = (size_t)BATCH * SEQ * SEQ;
    size_t i = (size_t)blockIdx.x * blockDim.x + threadIdx.x;
    if (i >= N) return;
    const int t = i & (SEQ - 1);
    size_t r = i >> 9;
    const int s = r & (SEQ - 1);
    const int b = (int)(r >> 9);
    float acc = 0.f;
#pragma unroll
    for (int h = 0; h < NUM_HEADS; ++h)
        acc += (float)attn[((size_t)(b * NUM_HEADS + h) * SEQ + s) * SEQ + t];
    out[i] = acc * (1.f / NUM_HEADS);
}

// ---------------------------------------------------------------------------
extern "C" void kernel_launch(void* const* d_in, const int* in_sizes, int n_in,
                              void* d_out, int out_size, void* d_ws, size_t ws_size,
                              hipStream_t stream)
{
    const float* q_in  = (const float*)d_in[0];
    const float* k_in  = (const float*)d_in[1];
    const float* v_in  = (const float*)d_in[2];
    const float* wq_w  = (const float*)d_in[3];
    const float* wq_b  = (const float*)d_in[4];
    const float* wk_w  = (const float*)d_in[5];
    const float* wk_b  = (const float*)d_in[6];
    const float* wv_w  = (const float*)d_in[7];
    const float* wv_b  = (const float*)d_in[8];
    const float* wo_w  = (const float*)d_in[9];
    const float* wo_b  = (const float*)d_in[10];
    const float* rbias = (const float*)d_in[11];

    const size_t MK  = (size_t)MROWS * D_MODEL;     // 8.4M elems
    const size_t WK  = (size_t)D_MODEL * D_MODEL;   // 1M elems
    const size_t ANN = (size_t)BATCH * NUM_HEADS * SEQ * SEQ; // 67M elems

    char* p = (char*)d_ws;
    h16* hXq = (h16*)p; p += MK * 2;    // later reused as V-projection output
    h16* hXk = (h16*)p; p += MK * 2;    // later reused as V^T
    h16* hXv = (h16*)p; p += MK * 2;    // later reused as ctx
    h16* hWq = (h16*)p; p += WK * 2;
    h16* hWk = (h16*)p; p += WK * 2;
    h16* hWv = (h16*)p; p += WK * 2;
    h16* hWo = (h16*)p; p += WK * 2;
    h16* hQ  = (h16*)p; p += MK * 2;
    h16* hK  = (h16*)p; p += MK * 2;
    h16* hAt = (h16*)p; p += ANN * 2;
    h16* hVp  = hXq;                    // V projection output (Xq dead after Q proj)
    h16* hVt  = hXk;                    // transposed V       (Xk dead after K proj)
    h16* hCtx = hXv;                    // context            (Xv dead after V proj)

    float* out_main = (float*)d_out;
    float* out_mean = out_main + MK;

    // 1. fp32 -> f16 conversions
    cvt_f32_to_f16<<<4096, 256, 0, stream>>>(q_in, hXq, (int)MK);
    cvt_f32_to_f16<<<4096, 256, 0, stream>>>(k_in, hXk, (int)MK);
    cvt_f32_to_f16<<<4096, 256, 0, stream>>>(v_in, hXv, (int)MK);
    cvt_f32_to_f16<<<2048, 256, 0, stream>>>(wq_w, hWq, (int)WK);
    cvt_f32_to_f16<<<2048, 256, 0, stream>>>(wk_w, hWk, (int)WK);
    cvt_f32_to_f16<<<2048, 256, 0, stream>>>(wv_w, hWv, (int)WK);
    cvt_f32_to_f16<<<2048, 256, 0, stream>>>(wo_w, hWo, (int)WK);

    // 2. Q/K/V projections (f16 out)
    dim3 ggrid(MROWS / 64, D_MODEL / 256);
    gemm_xwt<<<ggrid, 256, 0, stream>>>(hXq, hWq, wq_b, hQ,  nullptr, MROWS, D_MODEL, D_MODEL);
    gemm_xwt<<<ggrid, 256, 0, stream>>>(hXk, hWk, wk_b, hK,  nullptr, MROWS, D_MODEL, D_MODEL);
    gemm_xwt<<<ggrid, 256, 0, stream>>>(hXv, hWv, wv_b, hVp, nullptr, MROWS, D_MODEL, D_MODEL);

    // 3. V transpose per head
    transpose_v<<<(unsigned)((MK + 255) / 256), 256, 0, stream>>>(hVp, hVt);

    // 4. fused scores -> softmax -> attn f16 -> ctx
    attn_block<<<BATCH * NUM_HEADS * (SEQ / 16), 256, 0, stream>>>(hQ, hK, hVt, rbias, hAt, hCtx);

    // 5. output projection (fp32 out, region 0)
    gemm_xwt<<<ggrid, 256, 0, stream>>>(hCtx, hWo, wo_b, nullptr, out_main, MROWS, D_MODEL, D_MODEL);

    // 6. head-mean of attention (fp32, region 1)
    attn_mean<<<(unsigned)((ANN / NUM_HEADS + 255) / 256), 256, 0, stream>>>(hAt, out_mean);
}
